// ModulatedConv2d_52956946760301
// MI455X (gfx1250) — matrix operand order repaired and checked by reading, not compiled, not run on hardware
//
#include <hip/hip_runtime.h>

typedef __attribute__((ext_vector_type(16))) _Float16 v16h;
typedef __attribute__((ext_vector_type(8)))  float    v8f;
typedef __attribute__((ext_vector_type(4)))  float    v4f;

#define B_    8
#define CIN_  512
#define COUT_ 512
#define SD_   512
#define NCLS_ 35
#define H_    64
#define W_    64
#define KK_   9
#define CIPAD 40   // LDS row stride in halves (80B: 16B aligned, bank-spread)

// ---------------- s[b][ci] = style @ (mod_w^T / sqrt(SD)) + mod_b ----------------
__global__ void k_style(const float* __restrict__ style,
                        const float* __restrict__ mod_w,
                        const float* __restrict__ mod_b,
                        float* __restrict__ s_out) {
  int idx = blockIdx.x * 256 + threadIdx.x;            // b*CIN + ci
  if (idx >= B_ * CIN_) return;
  int b = idx / CIN_, ci = idx - b * CIN_;
  const float sc = 0.044194173824159216f;              // 1/sqrt(512)
  float acc = 0.f;
  for (int k = 0; k < SD_; ++k)
    acc += style[b * SD_ + k] * mod_w[ci * SD_ + k];
  s_out[idx] = acc * sc + mod_b[ci];
}

// ---------------- demod[b][co] = rsqrt(sum (scale*w*s)^2 + 1e-8) ----------------
__global__ __launch_bounds__(256) void k_demod(const float* __restrict__ weight,
                                               const float* __restrict__ s_in,
                                               float* __restrict__ demod) {
  __shared__ float red[256];
  int bc = blockIdx.x;
  int b = bc / COUT_, co = bc - b * COUT_;
  const float sc = 0.014731391274719738f;              // 1/sqrt(4608)
  float acc = 0.f;
  for (int i = threadIdx.x; i < CIN_ * KK_; i += 256) {
    int ci = i / KK_, tap = i - ci * KK_;
    float wv = sc * weight[(co * CIN_ + ci) * KK_ + tap] * s_in[b * CIN_ + ci];
    acc += wv * wv;
  }
  red[threadIdx.x] = acc;
  __syncthreads();
  for (int off = 128; off > 0; off >>= 1) {
    if (threadIdx.x < off) red[threadIdx.x] += red[threadIdx.x + off];
    __syncthreads();
  }
  if (threadIdx.x == 0) demod[bc] = rsqrtf(red[0] + 1e-8f);
}

// ---- f16 weights pre-swizzled into WMMA A-fragment order -----------------------
// layout: [b][ciChunk(16)][tap(9)][coChunk(32)][lane(32)][half(16)]
//   -> tap stride = 32KB, coChunk stride = 1KB: all A loads become
//      base + constant immediate offsets inside the conv kernel.
// A frag k order used here (contiguous): lane<16: M=lane, K(v,half) = 2v+half
//                     lane>=16: M=lane-16, K += 16
__global__ void k_wfrag(const float* __restrict__ weight,
                        const float* __restrict__ s_in,
                        const float* __restrict__ demod,
                        _Float16* __restrict__ wfrag) {
  long long idx = (long long)blockIdx.x * 256 + threadIdx.x;
  int hh      = (int)(idx & 15);
  int lane    = (int)((idx >> 4) & 31);
  int coChunk = (int)((idx >> 9) & 31);
  int rest    = (int)(idx >> 14);                      // (b*16+cc)*9 + tap
  int tap = rest % 9;
  int q   = rest / 9;
  int cc  = q & 15;
  int b   = q >> 4;
  int co  = coChunk * 16 + (lane & 15);
  int vg = hh >> 1, hl = hh & 1;
  int kbase = (vg < 4) ? (2 * vg + hl) : (16 + 2 * (vg - 4) + hl);
  int K = hh + ((lane >= 16) ? 16 : 0);
  int ci = cc * 32 + K;
  const float sc = 0.014731391274719738f;
  float val = sc * weight[(co * CIN_ + ci) * KK_ + tap]
                 * s_in[b * CIN_ + ci] * demod[b * COUT_ + co];
  wfrag[idx] = (_Float16)val;
}

// ---------------- main conv: implicit GEMM on v_wmma_f32_16x16x32_f16 -----------
// grid (H, COUT/256, B), 256 threads = 8 waves; wave = 64co x 32pix tile.
// Double-buffered LDS slab; next chunk's input is prefetched into registers
// (6 x global_load_b128 per thread) before the tap loop so global latency
// overlaps the 72 WMMAs of the current chunk.
__global__ __launch_bounds__(256) void k_conv(const float* __restrict__ xin,
                                              const _Float16* __restrict__ wfrag,
                                              float* __restrict__ conv) {
  const int h      = blockIdx.x;
  const int coTile = blockIdx.y;     // 0..1, 256 couts each
  const int b      = blockIdx.z;
  const int tid  = threadIdx.x;
  const int wave = tid >> 5;
  const int lane = tid & 31;
  const int waveM = wave & 3;        // 4 groups of 64 couts
  const int waveN = wave >> 2;       // 2 groups of 32 pixels
  const int l16 = lane & 15;
  const int lhi = lane >> 4;

  __shared__ _Float16 slab[2][198 * CIPAD];   // [r*66+c][ci], halo rows h-1..h+1

  // Horizontal halo (c=0 and c=65 <=> w=-1,64) is ALWAYS zero padding:
  // zero it once per buffer; staging below only writes core columns 1..64.
  for (int i = tid; i < 2 * 3 * 2 * 32; i += 256) {     // buf x r x side x ci
    int ci = i & 31;
    int t = i >> 5;                                     // 0..11
    int side = t & 1;
    int r = (t >> 1) % 3;
    int buf = (t >> 1) / 3;
    slab[buf][(r * 66 + (side ? 65 : 0)) * CIPAD + ci] = (_Float16)0.f;
  }

  v8f acc[4][2];
#pragma unroll
  for (int i = 0; i < 4; ++i)
#pragma unroll
    for (int j = 0; j < 2; ++j) {
      v8f z = {0.f, 0.f, 0.f, 0.f, 0.f, 0.f, 0.f, 0.f};
      acc[i][j] = z;
    }

  // staging geometry: core = 3 rows x 32 ci x 64 cols = 1536 float4 = 6/thread
  // f4idx = tid + k*256 : c4 = (f4idx&15)*4 ; ci = (f4idx>>4)&31 ; r = f4idx>>9
  v4f pf[6];

  // ---- prologue: stage chunk 0 into buffer 0 ----
#pragma unroll
  for (int k = 0; k < 6; ++k) {
    int f4 = tid + k * 256;
    int c4 = (f4 & 15) * 4, ci = (f4 >> 4) & 31, r = f4 >> 9;
    int srow = h + r - 1;
    v4f z = {0.f, 0.f, 0.f, 0.f};
    pf[k] = ((unsigned)srow < 64u)
          ? *(const v4f*)(xin + (((size_t)b * CIN_ + ci) << 12) + (srow << 6) + c4)
          : z;
  }
#pragma unroll
  for (int k = 0; k < 6; ++k) {
    int f4 = tid + k * 256;
    int c4 = (f4 & 15) * 4, ci = (f4 >> 4) & 31, r = f4 >> 9;
#pragma unroll
    for (int j = 0; j < 4; ++j)
      slab[0][(r * 66 + c4 + j + 1) * CIPAD + ci] = (_Float16)pf[k][j];
  }
  __syncthreads();

  // per-wave fragment-buffer half-offset (lane-dependent part is fixed)
  const unsigned laneOffH =
      (unsigned)(coTile * 16 + waveM * 4) * 512u + (unsigned)lane * 16u;

  for (int cc = 0; cc < 16; ++cc) {
    // issue prefetch of chunk cc+1 (loads in flight during WMMAs)
    if (cc < 15) {
#pragma unroll
      for (int k = 0; k < 6; ++k) {
        int f4 = tid + k * 256;
        int c4 = (f4 & 15) * 4, ci = (f4 >> 4) & 31, r = f4 >> 9;
        int srow = h + r - 1;
        v4f z = {0.f, 0.f, 0.f, 0.f};
        pf[k] = ((unsigned)srow < 64u)
              ? *(const v4f*)(xin + (((size_t)b * CIN_ + (cc + 1) * 32 + ci) << 12)
                              + (srow << 6) + c4)
              : z;
      }
    }

    // one base address per cc; taps/mf are constant immediate offsets
    const _Float16* abase = wfrag
        + (size_t)((unsigned)(b * 16 + cc) * 9u) * 16384u + laneOffH;
    const _Float16* sb = slab[cc & 1];

    for (int tap = 0; tap < 9; ++tap) {
      const int dh = tap / 3;
      const int dw = tap - dh * 3;
      v16h afr[4], bfr[2];
#pragma unroll
      for (int mf = 0; mf < 4; ++mf)
        afr[mf] = *(const v16h*)(abase + tap * 16384 + mf * 512);
#pragma unroll
      for (int nf = 0; nf < 2; ++nf) {
        int n = waveN * 32 + nf * 16 + l16;             // B col = lane%16
        // B frag k order used here (contiguous, same as A): lanes<16 hold K=0..15, lanes>=16 hold K=16..31
        bfr[nf] = *(const v16h*)(&sb[(dh * 66 + n + dw) * CIPAD + lhi * 16]);
      }
#pragma unroll
      for (int mf = 0; mf < 4; ++mf)
#pragma unroll
        for (int nf = 0; nf < 2; ++nf)
          acc[mf][nf] = __builtin_amdgcn_wmma_f32_16x16x32_f16(
              false, afr[mf], false, bfr[nf], (short)0, acc[mf][nf], false, false);
    }

    // commit prefetched chunk cc+1 into the other buffer
    if (cc < 15) {
#pragma unroll
      for (int k = 0; k < 6; ++k) {
        int f4 = tid + k * 256;
        int c4 = (f4 & 15) * 4, ci = (f4 >> 4) & 31, r = f4 >> 9;
#pragma unroll
        for (int j = 0; j < 4; ++j)
          slab[(cc + 1) & 1][(r * 66 + c4 + j + 1) * CIPAD + ci] = (_Float16)pf[k][j];
      }
    }
    __syncthreads();
  }

  // C/D 16x16 f32 layout: lane<16 -> M=r, N=lane ; lane>=16 -> M=r+8, N=lane-16
#pragma unroll
  for (int mf = 0; mf < 4; ++mf)
#pragma unroll
    for (int nf = 0; nf < 2; ++nf) {
      int w2 = waveN * 32 + nf * 16 + l16;
#pragma unroll
      for (int r = 0; r < 8; ++r) {
        int co = coTile * 256 + waveM * 64 + mf * 16 + lhi * 8 + r;
        conv[(((size_t)b * COUT_ + co) << 12) + (h << 6) + w2] = acc[mf][nf][r];
      }
    }
}

// ---------------- instance-norm stats per (b,co) --------------------------------
__global__ __launch_bounds__(256) void k_norm(const float* __restrict__ conv,
                                              float* __restrict__ muv,
                                              float* __restrict__ rstdv) {
  __shared__ float r1[256], r2[256];
  int bc = blockIdx.x;
  const float* p = conv + (size_t)bc * H_ * W_;
  float s = 0.f, s2 = 0.f;
  for (int i = threadIdx.x; i < H_ * W_; i += 256) {
    float v = p[i];
    s += v; s2 += v * v;
  }
  r1[threadIdx.x] = s; r2[threadIdx.x] = s2;
  __syncthreads();
  for (int off = 128; off > 0; off >>= 1) {
    if (threadIdx.x < off) {
      r1[threadIdx.x] += r1[threadIdx.x + off];
      r2[threadIdx.x] += r2[threadIdx.x + off];
    }
    __syncthreads();
  }
  if (threadIdx.x == 0) {
    float m = r1[0] * (1.f / 4096.f);
    float var = r2[0] * (1.f / 4096.f) - m * m;
    muv[bc] = m;
    rstdv[bc] = rsqrtf(var + 1e-5f);
  }
}

// ---------------- CLADE per-class affine tables ---------------------------------
__global__ void k_clade(const float* __restrict__ class_style,
                        const float* __restrict__ cw_w, const float* __restrict__ cw_b,
                        const float* __restrict__ cb_w, const float* __restrict__ cb_b,
                        float* __restrict__ cwv, float* __restrict__ cbv) {
  int idx = blockIdx.x * 256 + threadIdx.x;             // 2*NCLS*COUT
  int which = idx / (NCLS_ * COUT_);
  int j = idx - which * (NCLS_ * COUT_);
  int cls = j / COUT_, co = j - cls * COUT_;
  const float* wm = which ? cb_w : cw_w;
  const float sc = 0.044194173824159216f;               // 1/sqrt(512)
  float acc = 0.f;
  for (int k = 0; k < SD_; ++k)
    acc += class_style[cls * SD_ + k] * wm[co * SD_ + k];
  acc = acc * sc + (which ? cb_b[co] : cw_b[co]);
  if (which) cbv[j] = acc; else cwv[j] = acc;
}

// ---------------- normalize + CLADE gather/apply --------------------------------
__global__ void k_final(const float* __restrict__ conv,
                        const float* __restrict__ muv,
                        const float* __restrict__ rstdv,
                        const float* __restrict__ cwv,
                        const float* __restrict__ cbv,
                        const int* __restrict__ label,
                        float* __restrict__ out) {
  unsigned idx = blockIdx.x * 256u + threadIdx.x;       // [b][co][h][w] flat
  int w  = idx & 63;
  int h  = (idx >> 6) & 63;
  int co = (idx >> 12) & 511;
  int b  = idx >> 21;
  int cls = label[((b << 6) + h) * 64 + w];
  int bc = (b << 9) + co;
  float v = (conv[idx] - muv[bc]) * rstdv[bc];
  out[idx] = v * cwv[cls * COUT_ + co] + cbv[cls * COUT_ + co];
}

extern "C" void kernel_launch(void* const* d_in, const int* in_sizes, int n_in,
                              void* d_out, int out_size, void* d_ws, size_t ws_size,
                              hipStream_t stream) {
  const float* xin         = (const float*)d_in[0];
  const float* style       = (const float*)d_in[1];
  const float* class_style = (const float*)d_in[2];
  const float* weight      = (const float*)d_in[3];
  const float* mod_w       = (const float*)d_in[4];
  const float* mod_b       = (const float*)d_in[5];
  const float* cw_w        = (const float*)d_in[6];
  const float* cw_b        = (const float*)d_in[7];
  const float* cb_w        = (const float*)d_in[8];
  const float* cb_b        = (const float*)d_in[9];
  const int*   label       = (const int*)d_in[10];
  float* out = (float*)d_out;

  // workspace carve-out (all chunks 256B multiples)
  char* p = (char*)d_ws;
  float* s_mod = (float*)p;  p += B_ * CIN_ * 4;                      // 16 KB
  float* demod = (float*)p;  p += B_ * COUT_ * 4;                     // 16 KB
  float* muv   = (float*)p;  p += B_ * COUT_ * 4;                     // 16 KB
  float* rstdv = (float*)p;  p += B_ * COUT_ * 4;                     // 16 KB
  float* cwv   = (float*)p;  p += NCLS_ * COUT_ * 4;                  // 70 KB
  float* cbv   = (float*)p;  p += NCLS_ * COUT_ * 4;                  // 70 KB
  _Float16* wfrag = (_Float16*)p; p += (size_t)B_ * KK_ * CIN_ * COUT_ * 2;   // 37.75 MB
  float* conv  = (float*)p;  // 67.1 MB

  k_style<<<(B_ * CIN_ + 255) / 256, 256, 0, stream>>>(style, mod_w, mod_b, s_mod);
  k_demod<<<B_ * COUT_, 256, 0, stream>>>(weight, s_mod, demod);
  {
    long long total = (long long)B_ * KK_ * 16 * 32 * 32 * 16;        // 18,874,368
    k_wfrag<<<(unsigned)(total / 256), 256, 0, stream>>>(weight, s_mod, demod, wfrag);
  }
  {
    dim3 g(H_, COUT_ / 256, B_);
    k_conv<<<g, 256, 0, stream>>>(xin, wfrag, conv);
  }
  k_norm<<<B_ * COUT_, 256, 0, stream>>>(conv, muv, rstdv);
  k_clade<<<(2 * NCLS_ * COUT_) / 256, 256, 0, stream>>>(class_style, cw_w, cw_b,
                                                         cb_w, cb_b, cwv, cbv);
  k_final<<<(B_ * COUT_ * H_ * W_) / 256, 256, 0, stream>>>(conv, muv, rstdv,
                                                            cwv, cbv, label, out);
}
